// GAT_26714696581626
// MI455X (gfx1250) — matrix operand order repaired and checked by reading, not compiled, not run on hardware
//
#include <hip/hip_runtime.h>
#include <hip/hip_bf16.h>

// GAT (2-layer, PyG-style) for MI455X / gfx1250.
// Dense projections: v_wmma_f32_16x16x32_f16 with weights staged to LDS as
// f16 (zero-padded), so the k-loop has no guards/divergence. Edge softmax &
// scatter are f32 atomics -- all accumulators (<50MB) are L2-resident (192MB).

typedef __attribute__((ext_vector_type(16))) _Float16 v16h;
typedef __attribute__((ext_vector_type(8)))  float    v8f;

#define LRELU(e) ((e) > 0.0f ? (e) : 0.2f * (e))

// ---------------------------------------------------------------------------
// WMMA GEMM: C[M,NC] = A[M,KK] * B[KK,NC] (row-major f32). Compile-time
// KK/NC/NT so all address math folds to shifts + immediate offsets.
// NT = #16-col tiles (NT*16 >= NC). Block = 256 threads = 8 waves, each wave
// owns 16 rows and all NT column tiles (A fragment reused NT times / k-step).
//
// B^T staged in LDS as f16: lds[col*KLD + k], KLD = KK+16 (keeps each lane's
// 32B fragment read 32B-aligned). Cols >= NC are zero-filled, so the inner
// loop is guard-free and EXEC stays all-ones for the WMMAs.
//
// Fragment packing used here (wave32, contiguous k; not the ISA table order):
//   A 16x32 f16 : lanes 0-15 row=lane, halves[i]=K(k0+i);
//                 lanes 16-31 row=lane-16, halves[i]=K(k0+16+i).
//   B 32x16 f16 : lanes 0-15 col=lane, halves[i]=K(k0+i);
//                 lanes 16-31 col=lane-16, halves[i]=K(k0+16+i).
//   C/D 16x16 f32: lanes 0-15 col=lane rows 0-7; lanes 16-31 rows 8-15.
// ---------------------------------------------------------------------------
template <int NT, int KK, int NC>
__global__ void k_gemm_wmma(const float* __restrict__ A,
                            const float* __restrict__ B,
                            float* __restrict__ C,
                            int M) {
  constexpr int KLD = KK + 16;
  constexpr int NCP = NT * 16;
  __shared__ _Float16 lds[NCP * KLD];

  // stage B^T -> LDS (consecutive threads read consecutive cols: coalesced)
  for (int idx = threadIdx.x; idx < NCP * KK; idx += 256) {
    const int k = idx / NCP, col = idx - k * NCP;
    lds[col * KLD + k] =
        (col < NC) ? (_Float16)B[k * NC + col] : (_Float16)0.0f;
  }
  __syncthreads();

  const int wave = threadIdx.x >> 5;
  const int lane = threadIdx.x & 31;
  const int hs   = lane >> 4;                // half-select
  const int lr   = lane & 15;
  const int r0   = blockIdx.x * 128 + wave * 16;

  int arow = r0 + lr;
  if (arow >= M) arow = M - 1;               // clamp; stores row-guarded below
  const float* __restrict__ Ar = A + (size_t)arow * KK;

  v8f acc[NT];
#pragma unroll
  for (int t = 0; t < NT; ++t) acc[t] = (v8f){};

#pragma unroll
  for (int k0 = 0; k0 < KK; k0 += 32) {
    v16h a;
#pragma unroll
    for (int i = 0; i < 8; ++i) {
      a[i]     = (_Float16)Ar[k0 + hs * 16 + i];
      a[8 + i] = (_Float16)Ar[k0 + hs * 16 + 8 + i];
    }
    const int kb = k0 + hs * 16;
#pragma unroll
    for (int t = 0; t < NT; ++t) {
      const v16h b = *(const v16h*)&lds[(t * 16 + lr) * KLD + kb];
      acc[t] = __builtin_amdgcn_wmma_f32_16x16x32_f16(false, a, false, b,
                                                      (short)0, acc[t],
                                                      false, false);
    }
  }

  // epilogue: fast path for interior blocks (no row guards)
  float* __restrict__ Cp = C + (size_t)(r0 + hs * 8) * NC + lr;
  if (r0 + 16 <= M) {
#pragma unroll
    for (int t = 0; t < NT; ++t) {
      if (t * 16 + lr < NC) {
#pragma unroll
        for (int r = 0; r < 8; ++r) Cp[r * NC + t * 16] = acc[t][r];
      }
    }
  } else {
#pragma unroll
    for (int t = 0; t < NT; ++t) {
      if (t * 16 + lr < NC) {
#pragma unroll
        for (int r = 0; r < 8; ++r)
          if (r0 + hs * 8 + r < M) Cp[r * NC + t * 16] = acc[t][r];
      }
    }
  }
}

// ---------------------------------------------------------------------------
// Per-node attention halves: asrc[n,h] = dot(H[n,h,:], att_src[h,:]) etc.
// ---------------------------------------------------------------------------
__global__ void k_att_halves(const float* __restrict__ H,
                             const float* __restrict__ att_src,
                             const float* __restrict__ att_dst,
                             float* __restrict__ asrc,
                             float* __restrict__ adst,
                             int N, int heads, int ch) {
  const int i = blockIdx.x * blockDim.x + threadIdx.x;
  if (i >= N * heads) return;
  const int n = i / heads, h = i - n * heads;
  const float* __restrict__ hp = H + (size_t)n * heads * ch + (size_t)h * ch;
  float s1 = 0.f, s2 = 0.f;
  for (int c = 0; c < ch; ++c) {
    const float v = hp[c];
    s1 += v * att_src[h * ch + c];
    s2 += v * att_dst[h * ch + c];
  }
  asrc[i] = s1;
  adst[i] = s2;
}

// --------------------------- init kernels ----------------------------------
__global__ void k_fill_u32(unsigned* p, unsigned v, int n) {
  const int i = blockIdx.x * blockDim.x + threadIdx.x;
  if (i < n) p[i] = v;
}
__global__ void k_fill_f32(float* p, float v, int n) {
  const int i = blockIdx.x * blockDim.x + threadIdx.x;
  if (i < n) p[i] = v;
}
__global__ void k_bias_init(float* __restrict__ out,
                            const float* __restrict__ bias, int total, int C) {
  const int i = blockIdx.x * blockDim.x + threadIdx.x;
  if (i < total) out[i] = bias[i % C];
}

// ----------------------- monotonic float<->uint max -------------------------
__device__ __forceinline__ unsigned f32_enc(float f) {
  unsigned u = __float_as_uint(f);
  return (u & 0x80000000u) ? ~u : (u | 0x80000000u);
}
__device__ __forceinline__ float f32_dec(unsigned k) {
  unsigned u = (k & 0x80000000u) ? (k & 0x7fffffffu) : ~k;
  return __uint_as_float(u);
}

__device__ __forceinline__ void edge_sd(const int* __restrict__ src,
                                        const int* __restrict__ dst,
                                        const float* __restrict__ ew,
                                        int e, int E, int& s, int& d, float& w) {
  if (e < E) { s = src[e]; d = dst[e]; w = ew[e]; }
  else       { s = d = e - E; w = 1.0f; }          // self loop
}

// Pass 1: segment max of leaky-relu logits (encoded-uint atomicMax).
__global__ void k_edge_max(const int* __restrict__ src, const int* __restrict__ dst,
                           const float* __restrict__ ew,
                           const float* __restrict__ asrc, const float* __restrict__ adst,
                           unsigned* __restrict__ emax_u, int E, int ET, int heads) {
  const int e = blockIdx.x * blockDim.x + threadIdx.x;
  if (e >= ET) return;
  int s, d; float w;
  edge_sd(src, dst, ew, e, E, s, d, w);
  for (int h = 0; h < heads; ++h) {
    float lg = LRELU(asrc[(size_t)s * heads + h] + adst[(size_t)d * heads + h]);
    atomicMax(&emax_u[(size_t)d * heads + h], f32_enc(lg));
  }
}

__global__ void k_decode_max(float* emax, int n) {
  const int i = blockIdx.x * blockDim.x + threadIdx.x;
  if (i < n) emax[i] = f32_dec(((const unsigned*)emax)[i]);
}

// Pass 2: denom[d,h] += exp(e - emax[d,h])
__global__ void k_edge_expsum(const int* __restrict__ src, const int* __restrict__ dst,
                              const float* __restrict__ ew,
                              const float* __restrict__ asrc, const float* __restrict__ adst,
                              const float* __restrict__ emax, float* __restrict__ denom,
                              int E, int ET, int heads) {
  const int e = blockIdx.x * blockDim.x + threadIdx.x;
  if (e >= ET) return;
  int s, d; float w;
  edge_sd(src, dst, ew, e, E, s, d, w);
  for (int h = 0; h < heads; ++h) {
    float lg = LRELU(asrc[(size_t)s * heads + h] + adst[(size_t)d * heads + h]);
    atomicAdd(&denom[(size_t)d * heads + h], __expf(lg - emax[(size_t)d * heads + h]));
  }
}

// Pass 3: wave per edge; lane l covers channels l, l+32. agg[d,c] += H[s,c]*alpha.
__global__ void k_edge_agg(const int* __restrict__ src, const int* __restrict__ dst,
                           const float* __restrict__ ew,
                           const float* __restrict__ H,
                           const float* __restrict__ asrc, const float* __restrict__ adst,
                           const float* __restrict__ emax, const float* __restrict__ denom,
                           float* __restrict__ agg,
                           int E, int ET, int heads, int ch) {
  const int gid  = blockIdx.x * blockDim.x + threadIdx.x;
  const int e    = gid >> 5;
  const int lane = gid & 31;
  if (e >= ET) return;
  int s, d; float w;
  edge_sd(src, dst, ew, e, E, s, d, w);
  const int CH = heads * ch;
  for (int c = lane; c < CH; c += 32) {
    const int h = c / ch;
    float lg    = LRELU(asrc[(size_t)s * heads + h] + adst[(size_t)d * heads + h]);
    float ex    = __expf(lg - emax[(size_t)d * heads + h]);
    float alpha = ex / (denom[(size_t)d * heads + h] + 1e-16f) * w;
    atomicAdd(&agg[(size_t)d * CH + c], H[(size_t)s * CH + c] * alpha);
  }
}

__global__ void k_elu_inplace(float* p, int n) {
  const int i = blockIdx.x * blockDim.x + threadIdx.x;
  if (i < n) {
    const float v = p[i];
    p[i] = v > 0.f ? v : (__expf(v) - 1.0f);
  }
}

// ---------------------------------------------------------------------------
extern "C" void kernel_launch(void* const* d_in, const int* in_sizes, int n_in,
                              void* d_out, int out_size, void* d_ws, size_t ws_size,
                              hipStream_t stream) {
  const float* x   = (const float*)d_in[0];
  const int*   ei  = (const int*)  d_in[1];
  const float* ew  = (const float*)d_in[2];
  const float* W1  = (const float*)d_in[3];
  const float* as1 = (const float*)d_in[4];
  const float* ad1 = (const float*)d_in[5];
  const float* b1  = (const float*)d_in[6];
  const float* W2  = (const float*)d_in[7];
  const float* as2 = (const float*)d_in[8];
  const float* ad2 = (const float*)d_in[9];
  const float* b2  = (const float*)d_in[10];
  float* out = (float*)d_out;

  const int IN = 128, HEADS = 8, HID = 8, OUT = 40;
  const int CH1 = HEADS * HID;                 // 64
  const int N  = in_sizes[0] / IN;
  const int E  = in_sizes[1] / 2;
  const int ET = E + N;                        // edges incl. self loops
  const int* src = ei;
  const int* dst = ei + E;

  // workspace carve (floats): needs N*204*4 bytes (~82MB for N=100k)
  float* ws = (float*)d_ws;
  float* Hp1   = ws; ws += (size_t)N * CH1;
  float* agg1  = ws; ws += (size_t)N * CH1;    // becomes h1 after bias+ELU
  float* asr1  = ws; ws += (size_t)N * HEADS;
  float* adt1  = ws; ws += (size_t)N * HEADS;
  float* emax1 = ws; ws += (size_t)N * HEADS;
  float* den1  = ws; ws += (size_t)N * HEADS;
  float* Hp2   = ws; ws += (size_t)N * OUT;
  float* asr2  = ws; ws += (size_t)N;
  float* adt2  = ws; ws += (size_t)N;
  float* emax2 = ws; ws += (size_t)N;
  float* den2  = ws; ws += (size_t)N;

  const int B = 256;
  auto cdiv = [](long long a, long long b) { return (int)((a + b - 1) / b); };

  // ===== Layer 1: x[N,128] @ W1[128,64] =====
  k_gemm_wmma<4, 128, 64><<<cdiv(N, 128), B, 0, stream>>>(x, W1, Hp1, N);
  k_att_halves<<<cdiv((long long)N * HEADS, B), B, 0, stream>>>(Hp1, as1, ad1, asr1, adt1, N, HEADS, HID);
  k_fill_u32<<<cdiv((long long)N * HEADS, B), B, 0, stream>>>((unsigned*)emax1, 0u, N * HEADS);
  k_fill_f32<<<cdiv((long long)N * HEADS, B), B, 0, stream>>>(den1, 0.f, N * HEADS);
  k_bias_init<<<cdiv((long long)N * CH1, B), B, 0, stream>>>(agg1, b1, N * CH1, CH1);

  k_edge_max<<<cdiv(ET, B), B, 0, stream>>>(src, dst, ew, asr1, adt1, (unsigned*)emax1, E, ET, HEADS);
  k_decode_max<<<cdiv((long long)N * HEADS, B), B, 0, stream>>>(emax1, N * HEADS);
  k_edge_expsum<<<cdiv(ET, B), B, 0, stream>>>(src, dst, ew, asr1, adt1, emax1, den1, E, ET, HEADS);
  k_edge_agg<<<cdiv((long long)ET * 32, B), B, 0, stream>>>(src, dst, ew, Hp1, asr1, adt1, emax1, den1, agg1, E, ET, HEADS, HID);
  k_elu_inplace<<<cdiv((long long)N * CH1, B), B, 0, stream>>>(agg1, N * CH1);

  // ===== Layer 2: h1[N,64] @ W2[64,40] (heads=1, mean over 1 head = id) =====
  k_gemm_wmma<3, 64, 40><<<cdiv(N, 128), B, 0, stream>>>(agg1, W2, Hp2, N);
  k_att_halves<<<cdiv(N, B), B, 0, stream>>>(Hp2, as2, ad2, asr2, adt2, N, 1, OUT);
  k_fill_u32<<<cdiv(N, B), B, 0, stream>>>((unsigned*)emax2, 0u, N);
  k_fill_f32<<<cdiv(N, B), B, 0, stream>>>(den2, 0.f, N);
  k_bias_init<<<cdiv((long long)N * OUT, B), B, 0, stream>>>(out, b2, N * OUT, OUT);

  k_edge_max<<<cdiv(ET, B), B, 0, stream>>>(src, dst, ew, asr2, adt2, (unsigned*)emax2, E, ET, 1);
  k_decode_max<<<cdiv(N, B), B, 0, stream>>>(emax2, N);
  k_edge_expsum<<<cdiv(ET, B), B, 0, stream>>>(src, dst, ew, asr2, adt2, emax2, den2, E, ET, 1);
  k_edge_agg<<<cdiv((long long)ET * 32, B), B, 0, stream>>>(src, dst, ew, Hp2, asr2, adt2, emax2, den2, out, E, ET, 1, OUT);
}
